// RNNAttentionDecoder_22832046146360
// MI455X (gfx1250) — matrix operand order repaired and checked by reading, not compiled, not run on hardware
//
#include <hip/hip_runtime.h>

// ---------------------------------------------------------------------------
// Problem constants (from reference): B=16, S_DEC=64, S_ENC=128, H=1024,
// E=512, V=32000, L=2
// ---------------------------------------------------------------------------
#define BB   16
#define SD   64
#define SE   128
#define HH   1024
#define EE   512
#define VV   32000

// GEMM tiling
#define BM   128
#define BN   128
#define BK   32
#define LDSTR 40            // LDS row stride in elements (80 B): conflict-free frag reads
#define MPAD 128            // small-M GEMMs padded to this many rows

typedef __bf16 v16bf __attribute__((ext_vector_type(16)));
typedef __bf16 v8bf  __attribute__((ext_vector_type(8)));
typedef float  v8f   __attribute__((ext_vector_type(8)));

// ---------------------------------------------------------------------------
// LDS-tiled WMMA GEMM:  C[M,N] = A[M,K] * W[N,K]^T (+ bias[n])
//   A, W bf16 row-major; C f32.  Requirements: M%128==0, N%128==0, K%32==0.
//   permute==1 remaps output row m = s*BB+b -> b*SD+s (fused logits transpose).
//   Block: 256 threads = 8 waves (2 x 4). Block tile 128x128, K-step 32.
//   Each wave computes a 64x32 strip: 4x2 v_wmma_f32_16x16x32_bf16 accums.
//   Global K-slab for step k+1 is prefetched into registers while the WMMAs
//   consume the LDS tile for step k (single 20KB LDS buffer, 2 barriers/step).
// ---------------------------------------------------------------------------
__global__ __launch_bounds__(256)
void gemm_bf16_wmma(const __bf16* __restrict__ A,
                    const __bf16* __restrict__ W,
                    const float*  __restrict__ bias,
                    float*        __restrict__ C,
                    int M, int N, int K, int permute)
{
    __shared__ __bf16 lds[2 * BM * LDSTR];          // A tile then B tile (20 KB)
    __bf16* As = lds;
    __bf16* Bs = lds + BM * LDSTR;

    const int tid  = threadIdx.x;
    const int lane = tid & 31;
    const int wid  = tid >> 5;
    const int half = lane >> 4;                     // 16-lane half of the wave
    const int r    = lane & 15;

    const int wave_m = wid >> 2;                    // 0..1  -> 64 rows each
    const int wave_n = wid & 3;                     // 0..3  -> 32 cols each

    const int mblk = blockIdx.y * BM;
    const int nblk = blockIdx.x * BN;

    // cooperative tile load: thread covers rows (tid>>2, tid>>2+64),
    // 16-B chunk (tid&3)*8 within the 32-element K-slab
    const int lrow = tid >> 2;
    const int lcol = (tid & 3) * 8;

    const __bf16* agp  = A + (size_t)(mblk + lrow) * K + lcol;
    const __bf16* agp2 = agp + (size_t)64 * K;
    const __bf16* bgp  = W + (size_t)(nblk + lrow) * K + lcol;
    const __bf16* bgp2 = bgp + (size_t)64 * K;

    v8f acc[4][2];
#pragma unroll
    for (int i = 0; i < 4; ++i)
#pragma unroll
        for (int j = 0; j < 2; ++j)
#pragma unroll
            for (int v = 0; v < 8; ++v) acc[i][j][v] = 0.f;

    // prefetch K-slab 0
    v8bf a0 = *(const v8bf*)(agp);
    v8bf a1 = *(const v8bf*)(agp2);
    v8bf b0 = *(const v8bf*)(bgp);
    v8bf b1 = *(const v8bf*)(bgp2);

    for (int k = 0; k < K; k += BK) {
        if (k) __syncthreads();                     // prior frag reads complete
        *(v8bf*)(As + lrow * LDSTR + lcol)        = a0;
        *(v8bf*)(As + (lrow + 64) * LDSTR + lcol) = a1;
        *(v8bf*)(Bs + lrow * LDSTR + lcol)        = b0;
        *(v8bf*)(Bs + (lrow + 64) * LDSTR + lcol) = b1;
        __syncthreads();

        const int kn = k + BK;                      // prefetch next slab
        if (kn < K) {
            a0 = *(const v8bf*)(agp + kn);
            a1 = *(const v8bf*)(agp2 + kn);
            b0 = *(const v8bf*)(bgp + kn);
            b1 = *(const v8bf*)(bgp2 + kn);
        }

        // B fragments: lane holds col (n%16), K chunks at half*8 and 16+half*8
        v16bf bfr[2];
#pragma unroll
        for (int j = 0; j < 2; ++j) {
            const __bf16* p = Bs + (wave_n * 32 + j * 16 + r) * LDSTR + half * 8;
            v8bf lo = *(const v8bf*)(p);
            v8bf hi = *(const v8bf*)(p + 16);
#pragma unroll
            for (int i = 0; i < 8; ++i) { bfr[j][i] = lo[i]; bfr[j][i + 8] = hi[i]; }
        }

        // A fragments: lane holds row (m%16), K chunks at half*8 and 16+half*8
#pragma unroll
        for (int i = 0; i < 4; ++i) {
            const __bf16* p = As + (wave_m * 64 + i * 16 + r) * LDSTR + half * 8;
            v8bf lo = *(const v8bf*)(p);
            v8bf hi = *(const v8bf*)(p + 16);
            v16bf af;
#pragma unroll
            for (int q = 0; q < 8; ++q) { af[q] = lo[q]; af[q + 8] = hi[q]; }

            acc[i][0] = __builtin_amdgcn_wmma_f32_16x16x32_bf16(
                false, af, false, bfr[0], (short)0, acc[i][0], false, false);
            acc[i][1] = __builtin_amdgcn_wmma_f32_16x16x32_bf16(
                false, af, false, bfr[1], (short)0, acc[i][1], false, false);
        }
    }

    // D layout (ISA 7.12.2): VGPR v, lanes0-15 -> row +v, lanes16-31 -> row +v+8
#pragma unroll
    for (int i = 0; i < 4; ++i) {
        const int mrow = mblk + wave_m * 64 + i * 16 + half * 8;
#pragma unroll
        for (int j = 0; j < 2; ++j) {
            const int col = nblk + wave_n * 32 + j * 16 + r;
            const float bi = bias ? bias[col] : 0.f;
#pragma unroll
            for (int v = 0; v < 8; ++v) {
                const int row = mrow + v;
                const size_t orow = permute
                    ? ((size_t)(row & (BB - 1)) * SD + (row >> 4))
                    : (size_t)row;
                C[orow * (size_t)N + col] = acc[i][j][v] + bi;
            }
        }
    }
}

// ---------------------------------------------------------------------------
// f32 -> bf16 cast
// ---------------------------------------------------------------------------
__global__ __launch_bounds__(256)
void cast_f32_bf16(const float* __restrict__ in, __bf16* __restrict__ out, int n)
{
    int i = blockIdx.x * 256 + threadIdx.x;
    if (i < n) out[i] = (__bf16)in[i];
}

// cast [16,K] f32 -> [MPAD,K] bf16, zero padding rows 16..MPAD-1 (M%128==0)
__global__ __launch_bounds__(256)
void pad_rows_bf16(const float* __restrict__ in, __bf16* __restrict__ out, int K)
{
    int i = blockIdx.x * 256 + threadIdx.x;
    if (i >= MPAD * K) return;
    int row = i / K;
    out[i] = (row < 16) ? (__bf16)in[i] : (__bf16)0.0f;
}

// ---------------------------------------------------------------------------
// Attention: scores[b,s] = sum_h tanh(q[b,h] + k[b,s,h]) * wv[h]
// one wave per (b,s); shuffle reduction across 32 lanes
// ---------------------------------------------------------------------------
__global__ __launch_bounds__(256)
void attn_scores(const float* __restrict__ q, const float* __restrict__ kbuf,
                 const float* __restrict__ wv, float* __restrict__ scores)
{
    const int wid  = threadIdx.x >> 5;
    const int lane = threadIdx.x & 31;
    const int idx  = blockIdx.x * 8 + wid;          // 0 .. BB*SE-1
    const int b    = idx >> 7;
    const float* qp = q + (size_t)b * HH;
    const float* kp = kbuf + (size_t)idx * HH;
    float sum = 0.f;
    for (int h = lane; h < HH; h += 32)
        sum += tanhf(qp[h] + kp[h]) * wv[h];
#pragma unroll
    for (int off = 16; off > 0; off >>= 1)
        sum += __shfl_xor(sum, off, 32);
    if (lane == 0) scores[idx] = sum;
}

// masked softmax over S_ENC=128 per batch row
__global__ __launch_bounds__(128)
void attn_softmax(const float* __restrict__ scores, const int* __restrict__ vlen,
                  float* __restrict__ attn)
{
    __shared__ float sh[SE];
    const int b = blockIdx.x, s = threadIdx.x;
    const int len = vlen[b];
    float v = (s < len) ? scores[b * SE + s] : -1e6f;
    sh[s] = v;
    __syncthreads();
    for (int off = SE / 2; off > 0; off >>= 1) {
        if (s < off) sh[s] = fmaxf(sh[s], sh[s + off]);
        __syncthreads();
    }
    const float mx = sh[0];
    __syncthreads();
    const float e = __expf(v - mx);
    sh[s] = e;
    __syncthreads();
    for (int off = SE / 2; off > 0; off >>= 1) {
        if (s < off) sh[s] += sh[s + off];
        __syncthreads();
    }
    attn[b * SE + s] = e / sh[0];
}

// context[b,h] = sum_s attn[b,s] * enc[b,s,h]
__global__ __launch_bounds__(256)
void attn_context(const float* __restrict__ attn, const float* __restrict__ enc,
                  float* __restrict__ ctx)
{
    const int b = blockIdx.x;
    const int h = blockIdx.y * 256 + threadIdx.x;
    float c = 0.f;
    for (int s = 0; s < SE; ++s)
        c += attn[b * SE + s] * enc[((size_t)b * SE + s) * HH + h];
    ctx[(size_t)b * HH + h] = c;
}

// x0[m, :] = [context[b] (H) | emb[inputs[b,s]] (E)],  m = s*BB + b  (bf16)
__global__ __launch_bounds__(256)
void build_x0(const int* __restrict__ tok, const float* __restrict__ emb,
              const float* __restrict__ ctx, __bf16* __restrict__ x0)
{
    const int i = blockIdx.x * 256 + threadIdx.x;   // < SD*BB*(HH+EE)
    const int m = i / (HH + EE);
    const int c = i % (HH + EE);
    const int s = m >> 4, b = m & 15;
    float v;
    if (c < HH) v = ctx[(size_t)b * HH + c];
    else        v = emb[(size_t)tok[b * SD + s] * EE + (c - HH)];
    x0[i] = (__bf16)v;
}

// GRU gate fusion (torch order r,z,n):  h = (1-z)*tanh(i_n + r*hn) + z*h_prev
__global__ __launch_bounds__(256)
void gru_gates(const float* __restrict__ gi,   // [SD*BB, 3H] (no bias yet)
               const float* __restrict__ gh,   // [MPAD, 3H]  (rows 0..15 valid)
               const float* __restrict__ bih, const float* __restrict__ bhh,
               const float* __restrict__ hprev,  // [BB, H]
               float* __restrict__ hf, __bf16* __restrict__ hb)
{
    const int i = blockIdx.x * 256 + threadIdx.x;   // < SD*BB*HH
    const int m = i >> 10;
    const int h = i & (HH - 1);
    const int b = m & 15;
    const float* gim = gi + (size_t)m * (3 * HH);
    const float* ghb = gh + (size_t)b * (3 * HH);
    const float ir = gim[h]          + bih[h]          + ghb[h]          + bhh[h];
    const float iz = gim[HH + h]     + bih[HH + h]     + ghb[HH + h]     + bhh[HH + h];
    const float rr = 1.f / (1.f + __expf(-ir));
    const float zz = 1.f / (1.f + __expf(-iz));
    const float hn = ghb[2 * HH + h] + bhh[2 * HH + h];
    const float nn = tanhf(gim[2 * HH + h] + bih[2 * HH + h] + rr * hn);
    const float o  = (1.f - zz) * nn + zz * hprev[(size_t)b * HH + h];
    hf[i] = o;
    hb[i] = (__bf16)o;
}

// decoder_hidden tail: out[l,b,h] = h{l}[m = 63*16 + b, h]
__global__ __launch_bounds__(256)
void copy_hidden(const float* __restrict__ h1f, const float* __restrict__ h2f,
                 float* __restrict__ out)
{
    const int i = blockIdx.x * 256 + threadIdx.x;   // < 2*BB*HH
    if (i >= 2 * BB * HH) return;
    const int l = i >> 14;
    const int b = (i >> 10) & 15;
    const int h = i & (HH - 1);
    const float* src = l ? h2f : h1f;
    out[i] = src[((size_t)((SD - 1) * BB + b)) * HH + h];
}

// ---------------------------------------------------------------------------
// Host launcher
// ---------------------------------------------------------------------------
extern "C" void kernel_launch(void* const* d_in, const int* in_sizes, int n_in,
                              void* d_out, int out_size, void* d_ws, size_t ws_size,
                              hipStream_t stream)
{
    (void)in_sizes; (void)n_in; (void)out_size; (void)ws_size;

    const int*   inputs = (const int*)  d_in[0];
    const float* enc    = (const float*)d_in[1];   // [BB,SE,HH]
    const float* ehs    = (const float*)d_in[2];   // [2,BB,HH]
    const int*   vlens  = (const int*)  d_in[3];
    const float* emb    = (const float*)d_in[4];   // [VV,EE]
    const float* Wq     = (const float*)d_in[5];
    const float* Wk     = (const float*)d_in[6];
    const float* wv     = (const float*)d_in[7];
    const float* Wih0   = (const float*)d_in[8];
    const float* Whh0   = (const float*)d_in[9];
    const float* bih0   = (const float*)d_in[10];
    const float* bhh0   = (const float*)d_in[11];
    const float* Wih1   = (const float*)d_in[12];
    const float* Whh1   = (const float*)d_in[13];
    const float* bih1   = (const float*)d_in[14];
    const float* bhh1   = (const float*)d_in[15];
    const float* fcw    = (const float*)d_in[16];
    const float* fcb    = (const float*)d_in[17];
    float* out = (float*)d_out;                     // [BB,SD,VV] then [2,BB,HH]

    // ---- workspace bump allocator (256B aligned) ----
    char* wsb = (char*)d_ws;
    size_t off = 0;
    auto alloc = [&](size_t bytes) -> void* {
        void* p = wsb + off;
        off += (bytes + 255) & ~(size_t)255;
        return p;
    };

    const int M = SD * BB;           // 1024 rows (m = s*BB + b)
    __bf16* enc_bf   = (__bf16*)alloc((size_t)BB * SE * HH * 2);
    __bf16* Wk_bf    = (__bf16*)alloc((size_t)HH * HH * 2);
    __bf16* Wq_bf    = (__bf16*)alloc((size_t)HH * HH * 2);
    __bf16* ehs_pad  = (__bf16*)alloc((size_t)2 * MPAD * HH * 2);  // padded to 128 rows
    __bf16* Wih0_bf  = (__bf16*)alloc((size_t)3 * HH * (EE + HH) * 2);
    __bf16* Whh0_bf  = (__bf16*)alloc((size_t)3 * HH * HH * 2);
    __bf16* Wih1_bf  = (__bf16*)alloc((size_t)3 * HH * HH * 2);
    __bf16* Whh1_bf  = (__bf16*)alloc((size_t)3 * HH * HH * 2);
    __bf16* fcw_bf   = (__bf16*)alloc((size_t)VV * HH * 2);
    __bf16* x0_bf    = (__bf16*)alloc((size_t)M * (HH + EE) * 2);
    __bf16* h1_bf    = (__bf16*)alloc((size_t)M * HH * 2);
    __bf16* h2_bf    = (__bf16*)alloc((size_t)M * HH * 2);

    float* qbuf   = (float*)alloc((size_t)MPAD * HH * 4);
    float* kbuf   = (float*)alloc((size_t)BB * SE * HH * 4);
    float* scores = (float*)alloc((size_t)BB * SE * 4);
    float* attn   = (float*)alloc((size_t)BB * SE * 4);
    float* ctx    = (float*)alloc((size_t)BB * HH * 4);
    float* gh0    = (float*)alloc((size_t)MPAD * 3 * HH * 4);
    float* gh1    = (float*)alloc((size_t)MPAD * 3 * HH * 4);
    float* gi     = (float*)alloc((size_t)M * 3 * HH * 4);   // reused both layers
    float* h1f    = (float*)alloc((size_t)M * HH * 4);
    float* h2f    = (float*)alloc((size_t)M * HH * 4);

    auto cdiv = [](int a, int b) { return (a + b - 1) / b; };
    auto cast = [&](const float* src, __bf16* dst, int n) {
        cast_f32_bf16<<<cdiv(n, 256), 256, 0, stream>>>(src, dst, n);
    };

    // ---- 1) bf16 conversions ----
    cast(enc,  enc_bf,  BB * SE * HH);
    cast(Wk,   Wk_bf,   HH * HH);
    cast(Wq,   Wq_bf,   HH * HH);
    cast(Wih0, Wih0_bf, 3 * HH * (EE + HH));
    cast(Whh0, Whh0_bf, 3 * HH * HH);
    cast(Wih1, Wih1_bf, 3 * HH * HH);
    cast(Whh1, Whh1_bf, 3 * HH * HH);
    cast(fcw,  fcw_bf,  VV * HH);
    pad_rows_bf16<<<cdiv(MPAD * HH, 256), 256, 0, stream>>>(ehs,           ehs_pad,             HH);
    pad_rows_bf16<<<cdiv(MPAD * HH, 256), 256, 0, stream>>>(ehs + BB * HH, ehs_pad + MPAD * HH, HH);

    auto gemm = [&](const __bf16* A, const __bf16* W, const float* bias,
                    float* C, int m, int n, int k, int permute) {
        dim3 grid(n / BN, m / BM);
        gemm_bf16_wmma<<<grid, 256, 0, stream>>>(A, W, bias, C, m, n, k, permute);
    };

    // ---- 2) attention (context constant across decode steps) ----
    gemm(ehs_pad + MPAD * HH, Wq_bf, nullptr, qbuf, MPAD, HH, HH, 0);  // q = ehs[-1] @ Wq.T
    gemm(enc_bf,              Wk_bf, nullptr, kbuf, BB * SE, HH, HH, 0);
    attn_scores <<<BB * SE / 8, 256, 0, stream>>>(qbuf, kbuf, wv, scores);
    attn_softmax<<<BB, SE, 0, stream>>>(scores, vlens, attn);
    attn_context<<<dim3(BB, HH / 256), 256, 0, stream>>>(attn, enc, ctx);

    // ---- 3) x0 = [context | embedding], GRU layer 0 ----
    build_x0<<<cdiv(M * (HH + EE), 256), 256, 0, stream>>>(inputs, emb, ctx, x0_bf);
    gemm(ehs_pad, Whh0_bf, nullptr, gh0, MPAD, 3 * HH, HH, 0);
    gemm(x0_bf,   Wih0_bf, nullptr, gi,  M, 3 * HH, EE + HH, 0);
    gru_gates<<<cdiv(M * HH, 256), 256, 0, stream>>>(gi, gh0, bih0, bhh0, ehs, h1f, h1_bf);

    // ---- 4) GRU layer 1 ----
    gemm(ehs_pad + MPAD * HH, Whh1_bf, nullptr, gh1, MPAD, 3 * HH, HH, 0);
    gemm(h1_bf,               Wih1_bf, nullptr, gi,  M, 3 * HH, HH, 0);
    gru_gates<<<cdiv(M * HH, 256), 256, 0, stream>>>(gi, gh1, bih1, bhh1, ehs + BB * HH, h2f, h2_bf);

    // ---- 5) logits = h2 @ fc_w.T + fc_b, fused [s,b] -> [b,s] row permute ----
    gemm(h2_bf, fcw_bf, fcb, out, M, VV, HH, 1);

    // ---- 6) decoder hidden tail ----
    copy_hidden<<<cdiv(2 * BB * HH, 256), 256, 0, stream>>>(h1f, h2f,
                                                            out + (size_t)BB * SD * VV);
}
